// OutputBlock_53412213293605
// MI455X (gfx1250) — matrix operand order repaired and checked by reading, not compiled, not run on hardware
//
#include <hip/hip_runtime.h>
#include <hip/hip_bf16.h>
#include <math.h>

typedef __attribute__((ext_vector_type(16))) _Float16 v16h;
typedef __attribute__((ext_vector_type(8)))  _Float16 v8h;
typedef __attribute__((ext_vector_type(8)))  float    v8f;
typedef __attribute__((ext_vector_type(4)))  float    v4f;

constexpr int E_CNT   = 500000;
constexpr int CAT     = 276;     // feature dim
constexpr int CATP    = 288;     // padded to 18*16
constexpr int NRBF    = 20;
constexpr int KERBF   = 32;      // edge GEMM K padded to one 16x16x32 WMMA
constexpr int NATOMS  = 50000;
constexpr int NT      = 6;       // 6 N-tiles (96 cols) per wave; 3 groups = 288
constexpr int TAILV   = CAT - 17 * 16;  // = 4 valid lanes in the boundary tile

union AFrag { v16h v; v8h h[2]; };

// ---------------------------------------------------------------- utilities
__global__ void zero_f32_kernel(float* __restrict__ p, size_t n) {
  size_t i = (size_t)blockIdx.x * blockDim.x + threadIdx.x;
  size_t stride = (size_t)gridDim.x * blockDim.x;
  for (; i < n; i += stride) p[i] = 0.f;
}

// WT[n*kpad + k] = (k<ksrc && n<nsrc) ? (f16)W[k*nsrc + n] : 0   (transpose + pad)
__global__ void prep_wt_kernel(const float* __restrict__ W, _Float16* __restrict__ WT,
                               int ksrc, int nsrc, int kpad, int npad) {
  size_t total = (size_t)kpad * npad;
  size_t i = (size_t)blockIdx.x * blockDim.x + threadIdx.x;
  size_t stride = (size_t)gridDim.x * blockDim.x;
  for (; i < total; i += stride) {
    int n = (int)(i / kpad);
    int k = (int)(i % kpad);
    float v = (k < ksrc && n < nsrc) ? W[(size_t)k * nsrc + n] : 0.f;
    WT[i] = (_Float16)v;
  }
}

// node f32 [NATOMS][CAT] -> padded f16 activations [NATOMS][CATP]
__global__ void cvt_node_kernel(const float* __restrict__ nodef, _Float16* __restrict__ A16) {
  size_t total = (size_t)NATOMS * CATP;
  size_t i = (size_t)blockIdx.x * blockDim.x + threadIdx.x;
  size_t stride = (size_t)gridDim.x * blockDim.x;
  for (; i < total; i += stride) {
    int col = (int)(i % CATP);
    size_t row = i / CATP;
    float v = (col < CAT) ? nodef[row * CAT + col] : 0.f;
    A16[i] = (_Float16)v;
  }
}

// ------------------------------------------------------------- edge kernel
// One wave: 16(edge) x 96(col) patch of (e_rbf @ W_edge), gated by m_ji,
// scatter-added into nodef[dst]. PARTIAL => last tile covers the 276 boundary.
template <bool PARTIAL>
__global__ void __launch_bounds__(32)
edge_wmma_kernel(const float* __restrict__ m_ji,
                 const float* __restrict__ e_rbf,
                 const int*   __restrict__ nbr,         // [E][2]
                 const _Float16* __restrict__ WeT,      // [CATP][KERBF] transposed
                 float* __restrict__ nodef,             // [NATOMS][CAT] f32 accum
                 int ngbase) {
  const int lane = threadIdx.x;
  const int mt   = blockIdx.x;
  const int ng   = ngbase + blockIdx.y;
  const int half = lane >> 4;
  const int l16  = lane & 15;

  // ---- A fragment: e_rbf row (20 f32) -> f16, K padded to 32, branch-free
  const int mrow = mt * 16 + l16;
  const float* er = e_rbf + (size_t)mrow * NRBF;
  v4f f0 = *(const v4f*)(er + half * 8);       // K 0-3  | K 8-11
  v4f f1 = *(const v4f*)(er + half * 8 + 4);   // K 4-7  | K 12-15
  v4f f2 = *(const v4f*)(er + 16);             // K 16-19 (valid address for all lanes)
  v16h a;
#pragma unroll
  for (int i = 0; i < 4; ++i) {
    a[i]      = (_Float16)f0[i];
    a[4 + i]  = (_Float16)f1[i];
    a[8 + i]  = (half == 0) ? (_Float16)f2[i] : (_Float16)0.f;  // K16-19 | K24-27=0
    a[12 + i] = (_Float16)0.f;                                   // K20-23 | K28-31=0
  }

  v8f c[NT] = {};
#pragma unroll
  for (int t = 0; t < NT; ++t) {
    int n = ng * 96 + t * 16 + l16;
    AFrag bf;
    bf.h[0] = *(const v8h*)(WeT + (size_t)n * KERBF + half * 8);
    bf.h[1] = *(const v8h*)(WeT + (size_t)n * KERBF + half * 8 + 16);
    v16h b = bf.v;
    c[t] = __builtin_amdgcn_wmma_f32_16x16x32_f16(false, a, false, b,
                                                  (short)0, c[t], false, false);
  }

  // ---- hoist destination element offsets (32-bit mad, one nbr load per row)
  unsigned dofs[8];
#pragma unroll
  for (int r = 0; r < 8; ++r) {
    int dst = nbr[(size_t)(mt * 16 + r + half * 8) * 2];
    dofs[r] = (unsigned)dst * (unsigned)CAT;
  }

  // ---- epilogue: full tiles branch-free; boundary tile predicates once.
  // Batch the 8 m_ji loads per tile so they clause/overlap before the atomics.
  const unsigned mbase = (unsigned)(mt * 16 + half * 8) * (unsigned)CAT;
#pragma unroll
  for (int t = 0; t < NT; ++t) {
    const unsigned col = (unsigned)(ng * 96 + t * 16 + l16);
    if (!PARTIAL || t < NT - 1 || l16 < TAILV) {
      float mv[8];
#pragma unroll
      for (int r = 0; r < 8; ++r)
        mv[r] = m_ji[(size_t)(mbase + (unsigned)r * (unsigned)CAT + col)];
#pragma unroll
      for (int r = 0; r < 8; ++r)
        atomicAdd(nodef + (size_t)(dofs[r] + col), c[t][r] * mv[r]);
    }
  }
}

// --------------------------------------------------------------- MLP layer
template <bool PARTIAL, bool HAS_ACT, bool OUT32>
__global__ void __launch_bounds__(32)
mlp_wmma_kernel(const _Float16* __restrict__ A,    // [NATOMS][CATP]
                const _Float16* __restrict__ WT,   // [CATP][CATP] transposed
                const float* __restrict__ bias,    // [CAT] (used when HAS_ACT)
                _Float16* __restrict__ out16,      // [NATOMS][CATP] (when !OUT32)
                float* __restrict__ out32,         // [NATOMS][CAT]  (when OUT32)
                int ngbase) {
  const int lane = threadIdx.x;
  const int mt   = blockIdx.x;
  const int ng   = ngbase + blockIdx.y;
  const int half = lane >> 4;
  const int l16  = lane & 15;
  const int mrow = mt * 16 + l16;

  v8f c[NT] = {};
  for (int k0 = 0; k0 < CATP; k0 += 32) {
    // A: lanes 0-15 hold K k0+[0..7 | 16..23]; lanes 16-31 hold K k0+[8..15 | 24..31]
    const _Float16* ap = A + (size_t)mrow * CATP + k0 + half * 8;
    AFrag af;
    af.h[0] = *(const v8h*)(ap);
    af.h[1] = *(const v8h*)(ap + 16);
    v16h aa = af.v;

#pragma unroll
    for (int t = 0; t < NT; ++t) {
      int n = ng * 96 + t * 16 + l16;
      AFrag bf;
      bf.h[0] = *(const v8h*)(WT + (size_t)n * CATP + k0 + half * 8);
      bf.h[1] = *(const v8h*)(WT + (size_t)n * CATP + k0 + half * 8 + 16);
      v16h b = bf.v;
      c[t] = __builtin_amdgcn_wmma_f32_16x16x32_f16(false, aa, false, b,
                                                    (short)0, c[t], false, false);
    }
  }

  const unsigned rbase = (unsigned)(mt * 16 + half * 8);
#pragma unroll
  for (int t = 0; t < NT; ++t) {
    const unsigned col = (unsigned)(ng * 96 + t * 16 + l16);
    const bool valid = !PARTIAL || t < NT - 1 || l16 < TAILV;   // tile-uniform except boundary
    const float bb = (HAS_ACT && valid) ? bias[col] : 0.f;
#pragma unroll
    for (int r = 0; r < 8; ++r) {
      const unsigned row = rbase + (unsigned)r;
      float x = c[t][r];
      if (HAS_ACT) {
        x += bb;
        x = x / (1.f + __expf(-x));            // SiLU
      }
      if (OUT32) {
        if (valid) out32[(size_t)(row * (unsigned)CAT + col)] = x;
      } else {
        // always store: keeps padded cols (>=276) zeroed for the next layer
        out16[(size_t)(row * (unsigned)CATP + col)] = valid ? (_Float16)x : (_Float16)0.f;
      }
    }
  }
}

// ------------------------------------------------------------------ driver
extern "C" void kernel_launch(void* const* d_in, const int* in_sizes, int n_in,
                              void* d_out, int out_size, void* d_ws, size_t ws_size,
                              hipStream_t stream) {
  const float* m_ji    = (const float*)d_in[0];
  const float* e_rbf   = (const float*)d_in[1];
  const int*   nbr     = (const int*)d_in[2];
  // d_in[3] = num_atoms scalar (compile-time constant here)
  const float* W_edge  = (const float*)d_in[4];
  const float* W1      = (const float*)d_in[5];
  const float* b1      = (const float*)d_in[6];
  const float* W2      = (const float*)d_in[7];
  const float* b2      = (const float*)d_in[8];
  const float* W3      = (const float*)d_in[9];
  const float* b3      = (const float*)d_in[10];
  const float* W_final = (const float*)d_in[11];
  float* out = (float*)d_out;

  // workspace layout (all offsets 256B-aligned)
  char* ws = (char*)d_ws;
  float*     nodef = (float*)(ws);                               // 55,200,000 B
  _Float16*  A16a  = (_Float16*)(ws + 55200000);                 // 28,800,000 B
  _Float16*  A16b  = (_Float16*)(ws + 84000000);                 // 28,800,000 B
  _Float16*  WeT   = (_Float16*)(ws + 112800000);                //     18,432 B
  _Float16*  WT1   = (_Float16*)(ws + 112818432);                //    165,888 B
  _Float16*  WT2   = (_Float16*)(ws + 112984320);
  _Float16*  WT3   = (_Float16*)(ws + 113150208);
  _Float16*  WT4   = (_Float16*)(ws + 113316096);

  // 1) zero the f32 node accumulator
  {
    size_t n = (size_t)NATOMS * CAT;
    zero_f32_kernel<<<(int)((n + 255) / 256), 256, 0, stream>>>(nodef, n);
  }
  // 2) transpose + pad weights to f16
  prep_wt_kernel<<<(CATP * KERBF + 255) / 256, 256, 0, stream>>>(W_edge, WeT, NRBF, CAT, KERBF, CATP);
  prep_wt_kernel<<<(CATP * CATP + 255) / 256, 256, 0, stream>>>(W1, WT1, CAT, CAT, CATP, CATP);
  prep_wt_kernel<<<(CATP * CATP + 255) / 256, 256, 0, stream>>>(W2, WT2, CAT, CAT, CATP, CATP);
  prep_wt_kernel<<<(CATP * CATP + 255) / 256, 256, 0, stream>>>(W3, WT3, CAT, CAT, CATP, CATP);
  prep_wt_kernel<<<(CATP * CATP + 255) / 256, 256, 0, stream>>>(W_final, WT4, CAT, CAT, CATP, CATP);

  // 3) edge GEMM (WMMA) + gate + scatter-add: groups 0-1 full, group 2 partial
  edge_wmma_kernel<false><<<dim3(E_CNT / 16, 2), 32, 0, stream>>>(m_ji, e_rbf, nbr, WeT, nodef, 0);
  edge_wmma_kernel<true ><<<dim3(E_CNT / 16, 1), 32, 0, stream>>>(m_ji, e_rbf, nbr, WeT, nodef, 2);

  // 4) convert node features to padded f16 activations
  {
    size_t n = (size_t)NATOMS * CATP;
    cvt_node_kernel<<<(int)((n + 255) / 256), 256, 0, stream>>>(nodef, A16a);
  }

  // 5) MLP: 3 x (GEMM + bias + SiLU), then final GEMM -> d_out (f32)
  dim3 mf(NATOMS / 16, 2), mp(NATOMS / 16, 1);
  mlp_wmma_kernel<false, true, false><<<mf, 32, 0, stream>>>(A16a, WT1, b1, A16b, nullptr, 0);
  mlp_wmma_kernel<true,  true, false><<<mp, 32, 0, stream>>>(A16a, WT1, b1, A16b, nullptr, 2);
  mlp_wmma_kernel<false, true, false><<<mf, 32, 0, stream>>>(A16b, WT2, b2, A16a, nullptr, 0);
  mlp_wmma_kernel<true,  true, false><<<mp, 32, 0, stream>>>(A16b, WT2, b2, A16a, nullptr, 2);
  mlp_wmma_kernel<false, true, false><<<mf, 32, 0, stream>>>(A16a, WT3, b3, A16b, nullptr, 0);
  mlp_wmma_kernel<true,  true, false><<<mp, 32, 0, stream>>>(A16a, WT3, b3, A16b, nullptr, 2);
  mlp_wmma_kernel<false, false, true><<<mf, 32, 0, stream>>>(A16b, WT4, nullptr, nullptr, out, 0);
  mlp_wmma_kernel<true,  false, true><<<mp, 32, 0, stream>>>(A16b, WT4, nullptr, nullptr, out, 2);
}
